// FactorDecoder_71399536328779
// MI455X (gfx1250) — matrix operand order repaired and checked by reading, not compiled, not run on hardware
//
#include <hip/hip_runtime.h>
#include <hip/hip_bf16.h>
#include <math.h>

// ---------------------------------------------------------------------------
// FactorDecoder on MI455X (gfx1250): bf16 WMMA GEMMs + persistent GRU scan.
// ---------------------------------------------------------------------------

typedef __bf16 v8bf  __attribute__((ext_vector_type(8)));
typedef __bf16 v16bf __attribute__((ext_vector_type(16)));
typedef float  v8f   __attribute__((ext_vector_type(8)));

__device__ __forceinline__ v16bf cat8(v8bf lo, v8bf hi) {
    return __builtin_shufflevector(lo, hi, 0,1,2,3,4,5,6,7,8,9,10,11,12,13,14,15);
}
__device__ __forceinline__ float gelu_f(float x) {
    return 0.5f * x * (1.0f + erff(x * 0.70710678118654752f));
}
// Fast activations for the serial GRU chain: v_exp_f32 + v_rcp_f32 (and the
// gfx1250 native v_tanh_f32 when the builtin exists) instead of the IEEE
// division expansion (v_div_scale/v_rcp/Newton ~12 dependent VALU ops).
__device__ __forceinline__ float fast_sigmoid(float x) {
    return __builtin_amdgcn_rcpf(1.0f + __expf(-x));
}
__device__ __forceinline__ float fast_tanh(float x) {
#if __has_builtin(__builtin_amdgcn_tanhf)
    return __builtin_amdgcn_tanhf(x);
#elif __has_builtin(__builtin_amdgcn_tanh_f32)
    return __builtin_amdgcn_tanh_f32(x);
#else
    return 2.0f * fast_sigmoid(2.0f * x) - 1.0f;
#endif
}

// --------------------------- fp32 -> bf16 convert ---------------------------
__global__ void cvt_bf16_kernel(const float* __restrict__ s, __bf16* __restrict__ d, size_t n) {
    size_t i = (size_t)blockIdx.x * blockDim.x + threadIdx.x;
    size_t stride = (size_t)gridDim.x * blockDim.x;
    for (; i < n; i += stride) d[i] = (__bf16)s[i];
}

// ---------------- fp32 [R][C] -> bf16 [C][R] (B-operand-ready) --------------
__global__ void transpose_bf16_kernel(const float* __restrict__ s, __bf16* __restrict__ d,
                                      int R, int C) {
    size_t i = (size_t)blockIdx.x * blockDim.x + threadIdx.x;
    if (i < (size_t)R * C) {
        int r = (int)(i / C), c = (int)(i % C);
        d[(size_t)c * R + r] = (__bf16)s[i];
    }
}

// ------------------------------- WMMA GEMM ----------------------------------
// C[M,N] = act(A[M,Kd] @ W[Kd,N] + bias), W pre-transposed as Wt[N][Kd] bf16.
// Block = 128 threads = 4 waves; each wave owns NT 16x16 C tiles (one A-tile
// load feeds NT WMMAs -> NTx arithmetic intensity on the A side).
// Grid: (N / (64*NT), M/16).
template <int NT, int ACT_GELU, int OUT_BF16>
__global__ __launch_bounds__(128) void gemm_wmma_kernel(
    const __bf16* __restrict__ A, const __bf16* __restrict__ Wt,
    const float* __restrict__ bias, void* __restrict__ Cout,
    int M, int N, int Kd)
{
    const int wave = threadIdx.x >> 5;
    const int lane = threadIdx.x & 31;
    const int lane15 = lane & 15;
    const int half = lane >> 4;

    const int m0 = blockIdx.y << 4;
    const int n0 = (blockIdx.x * 4 + wave) * (NT * 16);

    const __bf16* ap = A + (size_t)(m0 + lane15) * Kd;       // lane = row M
    const __bf16* bp[NT];
#pragma unroll
    for (int j = 0; j < NT; ++j)
        bp[j] = Wt + (size_t)(n0 + j * 16 + lane15) * Kd;    // lane = col N

    v8f acc[NT];
#pragma unroll
    for (int j = 0; j < NT; ++j) acc[j] = (v8f){};

#pragma unroll 4
    for (int kb = 0; kb < Kd; kb += 32) {
        // A 16x32 bf16: half 0 -> K {0..7,16..23}, half 1 -> K {8..15,24..31}
        v16bf a = cat8(*(const v8bf*)(ap + kb + half * 8),
                       *(const v8bf*)(ap + kb + 16 + half * 8));
#pragma unroll
        for (int j = 0; j < NT; ++j) {
            // B 32x16 bf16: half 0 -> K {0..7,16..23}, half 1 -> K {8..15,24..31}
            v16bf b = cat8(*(const v8bf*)(bp[j] + kb + half * 8),
                           *(const v8bf*)(bp[j] + kb + 16 + half * 8));
            acc[j] = __builtin_amdgcn_wmma_f32_16x16x32_bf16(false, a, false, b,
                                                             (short)0, acc[j], false, false);
        }
    }

#pragma unroll
    for (int j = 0; j < NT; ++j) {
        const int col = n0 + j * 16 + lane15;
        const float bv = bias ? bias[col] : 0.0f;
#pragma unroll
        for (int i = 0; i < 8; ++i) {
            const int m = m0 + i + half * 8;     // C layout: VGPR i -> M=i (+8 hi half)
            float v = acc[j][i] + bv;
            if (ACT_GELU) v = gelu_f(v);
            if (OUT_BF16) ((__bf16*)Cout)[(size_t)m * N + col] = (__bf16)v;
            else          ((float*)Cout)[(size_t)m * N + col] = v;
        }
    }
}

// --------------------------- alpha = Ah @ Wa2 + ba2 -------------------------
__global__ __launch_bounds__(256) void alpha_gemv_kernel(
    const __bf16* __restrict__ Ah, const float* __restrict__ Wa2,
    const float* __restrict__ ba2, float* __restrict__ alpha, int rows)
{
    int r = blockIdx.x * blockDim.x + threadIdx.x;
    if (r >= rows) return;
    const __bf16* p = Ah + (size_t)r * 128;
    float s = 0.0f;
#pragma unroll 16
    for (int k = 0; k < 128; ++k) s += (float)p[k] * Wa2[k];
    alpha[r] = s + ba2[0];
}

// ------------------------------ GRU scan ------------------------------------
// 4 persistent workgroups (16 batch rows each), 256 threads = 8 waves.
// Wave w owns gate columns [16w,16w+16) of K=128; its B-operand tiles (r/z/n
// slices of W_hh, bf16) stay resident in VGPRs. h double-buffered in LDS,
// y-partials double-buffered -> one barrier per step. Fuses gelu + beta-dot
// + alpha; writes y_hat and hT.
__global__ __launch_bounds__(256) void gru_scan_kernel(
    const float* __restrict__ gx,     // [B,T,384] precomputed x@W_ih^T + b_ih
    const float* __restrict__ Whh,    // [384,128] fp32
    const float* __restrict__ b_hh,   // [384]
    const float* __restrict__ beta,   // [B,T,128]
    const float* __restrict__ alpha,  // [B,T]
    float* __restrict__ yhat,         // [B,T]
    float* __restrict__ hT,           // [B,128]
    int T)
{
    extern __shared__ unsigned char smem[];
    __bf16* sW  = (__bf16*)smem;            // [384*128] bf16   (96 KB)
    __bf16* sh0 = sW + 384 * 128;           // h ping          (4 KB)
    __bf16* sh1 = sh0 + 16 * 128;           // h pong          (4 KB)
    float*  sy  = (float*)(sh1 + 16 * 128); // [2][16] y partials (double-buffered)

    const int tid = threadIdx.x;
    const int b0  = blockIdx.x << 4;

    for (int i = tid; i < 384 * 128; i += 256) sW[i] = (__bf16)Whh[i];
    for (int i = tid; i < 16 * 128; i += 256) { sh0[i] = (__bf16)0.0f; sh1[i] = (__bf16)0.0f; }
    if (tid < 32) sy[tid] = 0.0f;
    __syncthreads();

    const int wave = tid >> 5, lane = tid & 31;
    const int lane15 = lane & 15, half = lane >> 4;
    const int col = (wave << 4) + lane15;   // this lane's hidden column (0..127)

    // Resident B tiles: gh = h @ W_hh^T  => B[k][n] = W_hh[n][k] (rows of W_hh).
    v16bf Bt[3][4];
#pragma unroll
    for (int g = 0; g < 3; ++g)
#pragma unroll
        for (int kb = 0; kb < 4; ++kb) {
            const __bf16* wp = sW + (size_t)(g * 128 + col) * 128 + kb * 32 + half * 8;
            Bt[g][kb] = cat8(*(const v8bf*)wp, *(const v8bf*)(wp + 16));
        }
    const float bhr = b_hh[col], bhz = b_hh[128 + col], bhn = b_hh[256 + col];

    float hold[8];
#pragma unroll
    for (int i = 0; i < 8; ++i) hold[i] = 0.0f;

    for (int t = 0; t < T; ++t) {
        const __bf16* shr = (t & 1) ? sh1 : sh0;
        __bf16*       shw = (t & 1) ? sh0 : sh1;
        float*        syc = sy + (t & 1) * 16;

        v16bf a[4];
#pragma unroll
        for (int kb = 0; kb < 4; ++kb) {
            const __bf16* apt = shr + lane15 * 128 + kb * 32;
            a[kb] = cat8(*(const v8bf*)(apt + half * 8),
                         *(const v8bf*)(apt + 16 + half * 8));
        }
        v8f ar = {}, az = {}, an = {};
#pragma unroll
        for (int kb = 0; kb < 4; ++kb) {
            ar = __builtin_amdgcn_wmma_f32_16x16x32_bf16(false, a[kb], false, Bt[0][kb], (short)0, ar, false, false);
            az = __builtin_amdgcn_wmma_f32_16x16x32_bf16(false, a[kb], false, Bt[1][kb], (short)0, az, false, false);
            an = __builtin_amdgcn_wmma_f32_16x16x32_bf16(false, a[kb], false, Bt[2][kb], (short)0, an, false, false);
        }

        float csum[8];
#pragma unroll
        for (int i = 0; i < 8; ++i) {
            const int m = i + half * 8;
            const int b = b0 + m;
            const size_t gbase = ((size_t)b * T + t) * 384 + col;
            const float xr = gx[gbase], xz = gx[gbase + 128], xn = gx[gbase + 256];
            const float r = fast_sigmoid(xr + ar[i] + bhr);
            const float z = fast_sigmoid(xz + az[i] + bhz);
            const float n = fast_tanh(xn + r * (an[i] + bhn));
            const float hv = (1.0f - z) * n + z * hold[i];
            hold[i] = hv;
            shw[m * 128 + col] = (__bf16)hv;
            csum[i] = gelu_f(hv) * beta[((size_t)b * T + t) * 128 + col];
        }
        if (t + 1 < T)
            __builtin_prefetch(gx + ((size_t)(b0 + half * 8) * T + t + 1) * 384 + col, 0, 1);

        // Reduce the beta-dot across the 16 columns of each lane-half, then
        // across waves via LDS atomics into the step's sy buffer.
#pragma unroll
        for (int i = 0; i < 8; ++i) {
            float c = csum[i];
#pragma unroll
            for (int mask = 1; mask < 16; mask <<= 1)
                c += __shfl_xor(c, mask, 32);
            if (lane15 == 0) atomicAdd(&syc[i + half * 8], c);
        }
        __syncthreads();   // h writes + sy adds visible; separates sy reuse by 2 steps
        if (tid < 16) {
            const size_t idx = (size_t)(b0 + tid) * T + t;
            yhat[idx] = syc[tid] + alpha[idx];
            syc[tid] = 0.0f;   // consumed; producers touch this buffer again only at t+2
        }
    }

#pragma unroll
    for (int i = 0; i < 8; ++i)
        hT[(size_t)(b0 + i + half * 8) * 128 + col] = hold[i];
}

// ------------------------------- launcher -----------------------------------
extern "C" void kernel_launch(void* const* d_in, const int* in_sizes, int n_in,
                              void* d_out, int out_size, void* d_ws, size_t ws_size,
                              hipStream_t stream) {
    (void)in_sizes; (void)n_in; (void)out_size; (void)ws_size;
    constexpr int B = 64, T = 512, F = 512, H = 512, K = 128;
    constexpr int BT = B * T;

    const float* firm  = (const float*)d_in[0];
    const float* inputs= (const float*)d_in[1];
    const float* Wa1   = (const float*)d_in[2];
    const float* ba1   = (const float*)d_in[3];
    const float* Wa2   = (const float*)d_in[4];
    const float* ba2   = (const float*)d_in[5];
    const float* Wb1   = (const float*)d_in[6];
    const float* bb1   = (const float*)d_in[7];
    const float* Wb2   = (const float*)d_in[8];
    const float* bb2   = (const float*)d_in[9];
    const float* Wf    = (const float*)d_in[10];
    const float* bfp   = (const float*)d_in[11];
    const float* W_ih  = (const float*)d_in[12];
    const float* W_hh  = (const float*)d_in[13];
    const float* b_ih  = (const float*)d_in[14];
    const float* b_hh  = (const float*)d_in[15];

    float* yhat = (float*)d_out;        // [B,T,1] flattened
    float* hTp  = yhat + (size_t)BT;    // [B,K]

    char* ws = (char*)d_ws;
    size_t off = 0;
    auto alloc = [&](size_t bytes) -> char* {
        char* p = ws + off;
        off = (off + bytes + 255) & ~(size_t)255;
        return p;
    };
    __bf16* fcB   = (__bf16*)alloc((size_t)BT * F * 2);
    __bf16* inB   = (__bf16*)alloc((size_t)BT * H * 2);
    __bf16* Ah    = (__bf16*)alloc((size_t)BT * K * 2);
    __bf16* Bh    = (__bf16*)alloc((size_t)BT * H * 2);
    __bf16* xB    = (__bf16*)alloc((size_t)BT * K * 2);
    float*  betaP = (float*)alloc((size_t)BT * K * 4);
    float*  gxP   = (float*)alloc((size_t)BT * 3 * K * 4);
    float*  alphaP= (float*)alloc((size_t)BT * 4);
    __bf16* Wa1t  = (__bf16*)alloc((size_t)K * F * 2);
    __bf16* Wb1t  = (__bf16*)alloc((size_t)H * F * 2);
    __bf16* Wb2t  = (__bf16*)alloc((size_t)K * H * 2);
    __bf16* Wft   = (__bf16*)alloc((size_t)K * H * 2);
    __bf16* WihB  = (__bf16*)alloc((size_t)3 * K * K * 2);

    // 1) Activation + weight precision conversion (bf16 operands, fp32 accum).
    cvt_bf16_kernel<<<4096, 256, 0, stream>>>(firm,   fcB, (size_t)BT * F);
    cvt_bf16_kernel<<<4096, 256, 0, stream>>>(inputs, inB, (size_t)BT * H);
    cvt_bf16_kernel<<<((3 * K * K) + 255) / 256, 256, 0, stream>>>(W_ih, WihB, (size_t)3 * K * K);
    transpose_bf16_kernel<<<((size_t)F * K + 255) / 256, 256, 0, stream>>>(Wa1, Wa1t, F, K);
    transpose_bf16_kernel<<<((size_t)F * H + 255) / 256, 256, 0, stream>>>(Wb1, Wb1t, F, H);
    transpose_bf16_kernel<<<((size_t)H * K + 255) / 256, 256, 0, stream>>>(Wb2, Wb2t, H, K);
    transpose_bf16_kernel<<<((size_t)H * K + 255) / 256, 256, 0, stream>>>(Wf,  Wft,  H, K);

    const dim3 blk(128);
    // 2) Ah = gelu(firm@Wa1 + ba1)            [BT,128] bf16   (NT=2: 1 block-col)
    gemm_wmma_kernel<2, 1, 1><<<dim3(1, BT / 16), blk, 0, stream>>>(fcB, Wa1t, ba1, Ah, BT, K, F);
    // 3) Bh = gelu(firm@Wb1 + bb1)            [BT,512] bf16   (NT=4: 2 block-cols)
    gemm_wmma_kernel<4, 1, 1><<<dim3(2, BT / 16), blk, 0, stream>>>(fcB, Wb1t, bb1, Bh, BT, H, F);
    // 4) x  = inputs@Wf + bf                  [BT,128] bf16
    gemm_wmma_kernel<2, 0, 1><<<dim3(1, BT / 16), blk, 0, stream>>>(inB, Wft, bfp, xB, BT, K, H);
    // 5) alpha = Ah@Wa2 + ba2                 [BT] fp32
    alpha_gemv_kernel<<<BT / 256, 256, 0, stream>>>(Ah, Wa2, ba2, alphaP, BT);
    // 6) beta = Bh@Wb2 + bb2                  [BT,128] fp32
    gemm_wmma_kernel<2, 0, 0><<<dim3(1, BT / 16), blk, 0, stream>>>(Bh, Wb2t, bb2, betaP, BT, K, H);
    // 7) gx = x@W_ih^T + b_ih                 [BT,384] fp32   (NT=3: 2 block-cols)
    gemm_wmma_kernel<3, 0, 0><<<dim3(2, BT / 16), blk, 0, stream>>>(xB, WihB, b_ih, gxP, BT, 3 * K, K);

    // 8) GRU scan + fused beta-dot/alpha; 4 persistent WGs, W_hh in LDS.
    const size_t smem = (size_t)384 * 128 * 2 + 2 * 16 * 128 * 2 + 2 * 16 * 4;
    gru_scan_kernel<<<4, 256, smem, stream>>>(gxP, W_hh, b_hh, betaP, alphaP, yhat, hTp, T);
}
